// EagerAttentionModel_44693429682907
// MI455X (gfx1250) — matrix operand order repaired and checked by reading, not compiled, not run on hardware
//
#include <hip/hip_runtime.h>
#include <hip/hip_bf16.h>

// ---------------------------------------------------------------------------
// EagerAttention on gfx1250: bf16 WMMA GEMMs + flash attention (online softmax)
// ---------------------------------------------------------------------------

typedef __attribute__((ext_vector_type(16))) __bf16 v16bf;
typedef __attribute__((ext_vector_type(8)))  float  v8f;
typedef __attribute__((ext_vector_type(4)))  unsigned int u32x4;
typedef __attribute__((ext_vector_type(4)))  float  f32x4;

#define WMMA_BF16(A, B, C) \
  __builtin_amdgcn_wmma_f32_16x16x32_bf16(false, (A), false, (B), (short)0, (C), false, false)

union AF {            // one 16x32 (A) or 32x16 (B) bf16 fragment: 8 VGPRs / lane
  v16bf v;
  u32x4 q[2];
};

__device__ __forceinline__ unsigned int f2bf(float f) {
  unsigned int u = __float_as_uint(f);
  u += 0x7fffu + ((u >> 16) & 1u);      // round-to-nearest-even
  return u >> 16;
}

// ---------------------------------------------------------------------------
// fp32 -> bf16 convert, 8 elements / thread, b128 packed stores
// ---------------------------------------------------------------------------
__global__ void cvt_bf16_kernel(const float* __restrict__ in,
                                unsigned int* __restrict__ out, int n8) {
  int i = blockIdx.x * blockDim.x + threadIdx.x;   // index in units of 8 floats
  if (i >= n8) return;
  f32x4 a = *(const f32x4*)(in + (size_t)i * 8);
  f32x4 b = *(const f32x4*)(in + (size_t)i * 8 + 4);
  u32x4 p;
  p.x = f2bf(a.x) | (f2bf(a.y) << 16);
  p.y = f2bf(a.z) | (f2bf(a.w) << 16);
  p.z = f2bf(b.x) | (f2bf(b.y) << 16);
  p.w = f2bf(b.z) | (f2bf(b.w) << 16);
  *(u32x4*)(out + (size_t)i * 4) = p;
}

// ---------------------------------------------------------------------------
// C[M,N] = A[M,K](bf16,row) x W[N,K](bf16,row, torch convention) + bias[N]
// MODE 0: bf16 out[M,N] * scale   (Q/K paths)
// MODE 1: bf16 out[N,M] * scale   (V path -> head-contiguous transpose,
//                                  packed 16B stores: 8 consecutive m / lane)
// MODE 2: f32  out[M,N]           (final projection)
// Block = 128 threads (4 waves), block tile 128x128, wave tile 64x64.
// ---------------------------------------------------------------------------
template <int MODE>
__global__ __launch_bounds__(128)
void gemm_bf16_kernel(const unsigned short* __restrict__ A,
                      const unsigned short* __restrict__ Bw,
                      const float* __restrict__ bias,
                      void* __restrict__ out,
                      int M, int N, int K, float scale) {
  const int lane = threadIdx.x & 31;
  const int wave = threadIdx.x >> 5;
  const int lmod = lane & 15;
  const int lhi  = lane >> 4;
  const int mBase = blockIdx.y * 128 + (wave >> 1) * 64;
  const int nBase = blockIdx.x * 128 + (wave & 1) * 64;

  v8f acc[4][4] = {};

  for (int kk = 0; kk < K; kk += 32) {
    AF a[4];
#pragma unroll
    for (int i = 0; i < 4; ++i) {
      const unsigned short* rp = A + (size_t)(mBase + i * 16 + lmod) * K + kk + lhi * 16;
      a[i].q[0] = *(const u32x4*)rp;
      a[i].q[1] = *(const u32x4*)(rp + 8);
    }
    AF b[4];
#pragma unroll
    for (int j = 0; j < 4; ++j) {
      const unsigned short* rp = Bw + (size_t)(nBase + j * 16 + lmod) * K + kk + lhi * 16;
      b[j].q[0] = *(const u32x4*)rp;
      b[j].q[1] = *(const u32x4*)(rp + 8);
    }
#pragma unroll
    for (int i = 0; i < 4; ++i)
#pragma unroll
      for (int j = 0; j < 4; ++j)
        acc[i][j] = WMMA_BF16(a[i].v, b[j].v, acc[i][j]);
  }

  float bj[4];
#pragma unroll
  for (int j = 0; j < 4; ++j) bj[j] = bias[nBase + j * 16 + lmod];

#pragma unroll
  for (int i = 0; i < 4; ++i) {
#pragma unroll
    for (int j = 0; j < 4; ++j) {
      const int n = nBase + j * 16 + lmod;
      if (MODE == 1) {
        // 8 consecutive m per lane -> single 16B packed bf16 store
        u32x4 p;
#pragma unroll
        for (int rr = 0; rr < 4; ++rr) {
          const unsigned int lo = f2bf((acc[i][j][2 * rr]     + bj[j]) * scale);
          const unsigned int hi = f2bf((acc[i][j][2 * rr + 1] + bj[j]) * scale);
          p[rr] = lo | (hi << 16);
        }
        const int m0 = mBase + i * 16 + lhi * 8;
        *(u32x4*)((unsigned short*)out + (size_t)n * M + m0) = p;
      } else {
#pragma unroll
        for (int r = 0; r < 8; ++r) {
          const int m = mBase + i * 16 + lhi * 8 + r;
          const float c = acc[i][j][r] + bj[j];
          if (MODE == 0) {
            ((unsigned short*)out)[(size_t)m * N + n] = (unsigned short)f2bf(c * scale);
          } else {
            ((float*)out)[(size_t)m * N + n] = c;
          }
        }
      }
    }
  }
}

// ---------------------------------------------------------------------------
// Causal flash attention, one head x 64 queries per block (4 waves x 16 rows).
// Q [S,H] bf16 (pre-scaled by 1/sqrt(HD)), K [S,H] bf16, Vt [H,S] bf16.
// Output O [S,H] bf16.
// ---------------------------------------------------------------------------
__global__ __launch_bounds__(128)
void flash_attn_kernel(const unsigned short* __restrict__ Q,
                       const unsigned short* __restrict__ Kb,
                       const unsigned short* __restrict__ Vt,
                       unsigned short* __restrict__ O,
                       int S, int H) {
  __shared__ unsigned short lds[4][16 * 64];   // per-wave private P tile

  const int lane = threadIdx.x & 31;
  const int wave = threadIdx.x >> 5;
  const int lmod = lane & 15;
  const int lhi  = lane >> 4;
  const int head = blockIdx.x;
  const int hOff = head * 64;
  const int qBase = blockIdx.y * 64 + wave * 16;

  // Resident Q fragments: rows qBase..qBase+15, d = 0..63 (two 16x32 A-frags)
  AF qf[2];
  {
    const unsigned short* rp = Q + (size_t)(qBase + lmod) * H + hOff;
    qf[0].q[0] = *(const u32x4*)(rp + lhi * 16);
    qf[0].q[1] = *(const u32x4*)(rp + 8 + lhi * 16);
    qf[1].q[0] = *(const u32x4*)(rp + 32 + lhi * 16);
    qf[1].q[1] = *(const u32x4*)(rp + 40 + lhi * 16);
  }

  v8f oacc[4] = {};
  float mrow[8], lrow[8];
#pragma unroll
  for (int r = 0; r < 8; ++r) { mrow[r] = -1e30f; lrow[r] = 0.0f; }

  const int nkt = (qBase + 79) / 64;           // key tiles covering k <= qBase+15

  for (int kt = 0; kt < nkt; ++kt) {
    const int kBase = kt * 64;

    // ---- S = Q x K^T (16 queries x 64 keys) : 8 WMMAs -------------------
    v8f s[4] = {};
#pragma unroll
    for (int j = 0; j < 4; ++j) {
      const unsigned short* rp =
          Kb + (size_t)(kBase + j * 16 + lmod) * H + hOff + lhi * 16;
      AF b0; b0.q[0] = *(const u32x4*)rp;        b0.q[1] = *(const u32x4*)(rp + 8);
      AF b1; b1.q[0] = *(const u32x4*)(rp + 32); b1.q[1] = *(const u32x4*)(rp + 40);
      s[j] = WMMA_BF16(qf[0].v, b0.v, s[j]);
      s[j] = WMMA_BF16(qf[1].v, b1.v, s[j]);
    }

    // ---- causal mask ----------------------------------------------------
    if (kBase + 63 > qBase) {
#pragma unroll
      for (int j = 0; j < 4; ++j) {
        const int kg = kBase + j * 16 + lmod;
#pragma unroll
        for (int r = 0; r < 8; ++r) {
          const int qg = qBase + lhi * 8 + r;
          if (kg > qg) s[j][r] = -1e30f;
        }
      }
    }

    // ---- online softmax -------------------------------------------------
    float corr[8];
#pragma unroll
    for (int r = 0; r < 8; ++r) {
      float mx = fmaxf(fmaxf(s[0][r], s[1][r]), fmaxf(s[2][r], s[3][r]));
      mx = fmaxf(mx, __shfl_xor(mx, 1, 16));
      mx = fmaxf(mx, __shfl_xor(mx, 2, 16));
      mx = fmaxf(mx, __shfl_xor(mx, 4, 16));
      mx = fmaxf(mx, __shfl_xor(mx, 8, 16));
      const float mnew = fmaxf(mrow[r], mx);
      corr[r] = __expf(mrow[r] - mnew);
      mrow[r] = mnew;
    }

    unsigned short* lw = lds[wave];
#pragma unroll
    for (int r = 0; r < 8; ++r) {
      float rs = 0.0f;
#pragma unroll
      for (int j = 0; j < 4; ++j) {
        const float p = __expf(s[j][r] - mrow[r]);
        rs += p;
        lw[(lhi * 8 + r) * 64 + j * 16 + lmod] = (unsigned short)f2bf(p);
      }
      rs += __shfl_xor(rs, 1, 16);
      rs += __shfl_xor(rs, 2, 16);
      rs += __shfl_xor(rs, 4, 16);
      rs += __shfl_xor(rs, 8, 16);
      lrow[r] = lrow[r] * corr[r] + rs;
#pragma unroll
      for (int j = 0; j < 4; ++j) oacc[j][r] *= corr[r];
    }

    // ---- reload P as A-fragments (same-wave DS ordering) ----------------
    AF pf[2];
    {
      const unsigned short* lr = lds[wave] + lmod * 64;
      pf[0].q[0] = *(const u32x4*)(lr + lhi * 16);
      pf[0].q[1] = *(const u32x4*)(lr + 8 + lhi * 16);
      pf[1].q[0] = *(const u32x4*)(lr + 32 + lhi * 16);
      pf[1].q[1] = *(const u32x4*)(lr + 40 + lhi * 16);
    }

    // ---- O += P x V : 8 WMMAs ------------------------------------------
#pragma unroll
    for (int j = 0; j < 4; ++j) {
      const unsigned short* vp =
          Vt + (size_t)(hOff + j * 16 + lmod) * S + kBase + lhi * 16;
      AF b0; b0.q[0] = *(const u32x4*)vp;        b0.q[1] = *(const u32x4*)(vp + 8);
      AF b1; b1.q[0] = *(const u32x4*)(vp + 32); b1.q[1] = *(const u32x4*)(vp + 40);
      oacc[j] = WMMA_BF16(pf[0].v, b0.v, oacc[j]);
      oacc[j] = WMMA_BF16(pf[1].v, b1.v, oacc[j]);
    }
  }

  // ---- finalize: O /= l, store bf16 [S,H] -------------------------------
#pragma unroll
  for (int r = 0; r < 8; ++r) {
    const float inv = 1.0f / lrow[r];
    unsigned short* orow = O + (size_t)(qBase + lhi * 8 + r) * H + hOff;
#pragma unroll
    for (int j = 0; j < 4; ++j)
      orow[j * 16 + lmod] = (unsigned short)f2bf(oacc[j][r] * inv);
  }
}

// ---------------------------------------------------------------------------
// Host-side orchestration
// ---------------------------------------------------------------------------
extern "C" void kernel_launch(void* const* d_in, const int* in_sizes, int n_in,
                              void* d_out, int out_size, void* d_ws, size_t ws_size,
                              hipStream_t stream) {
  (void)in_sizes; (void)n_in; (void)out_size; (void)ws_size;
  const int S = 4096, H = 1024, NHn = 16;
  const float SCALING = 0.125f;   // 1/sqrt(64)

  const float* x  = (const float*)d_in[0];
  const float* Wq = (const float*)d_in[1];
  const float* bq = (const float*)d_in[2];
  const float* Wk = (const float*)d_in[3];
  const float* bk = (const float*)d_in[4];
  const float* Wv = (const float*)d_in[5];
  const float* bv = (const float*)d_in[6];
  const float* Wo = (const float*)d_in[7];
  const float* bo = (const float*)d_in[8];

  char* ws = (char*)d_ws;
  const size_t MB = 1024 * 1024;
  unsigned short* xb  = (unsigned short*)(ws + 0 * MB);   //  8 MB  x  bf16 [S,H]
  unsigned short* Wqb = (unsigned short*)(ws + 8 * MB);   //  2 MB
  unsigned short* Wkb = (unsigned short*)(ws + 10 * MB);  //  2 MB
  unsigned short* Wvb = (unsigned short*)(ws + 12 * MB);  //  2 MB
  unsigned short* Wob = (unsigned short*)(ws + 14 * MB);  //  2 MB
  unsigned short* Qb  = (unsigned short*)(ws + 16 * MB);  //  8 MB  Q bf16 [S,H] (pre-scaled)
  unsigned short* Kbb = (unsigned short*)(ws + 24 * MB);  //  8 MB  K bf16 [S,H]
  unsigned short* Vt  = (unsigned short*)(ws + 32 * MB);  //  8 MB  V bf16 [H,S]
  unsigned short* Ab  = (unsigned short*)(ws + 40 * MB);  //  8 MB  attn out bf16 [S,H]

  // 1) fp32 -> bf16 conversions (8 elems / thread, vectorized)
  {
    const int nX8 = S * H / 8, nW8 = H * H / 8;
    cvt_bf16_kernel<<<(nX8 + 255) / 256, 256, 0, stream>>>(x,  (unsigned int*)xb,  nX8);
    cvt_bf16_kernel<<<(nW8 + 255) / 256, 256, 0, stream>>>(Wq, (unsigned int*)Wqb, nW8);
    cvt_bf16_kernel<<<(nW8 + 255) / 256, 256, 0, stream>>>(Wk, (unsigned int*)Wkb, nW8);
    cvt_bf16_kernel<<<(nW8 + 255) / 256, 256, 0, stream>>>(Wv, (unsigned int*)Wvb, nW8);
    cvt_bf16_kernel<<<(nW8 + 255) / 256, 256, 0, stream>>>(Wo, (unsigned int*)Wob, nW8);
  }

  // 2) QKV projections (WMMA bf16)
  dim3 gGrid(H / 128, S / 128), gBlk(128);
  gemm_bf16_kernel<0><<<gGrid, gBlk, 0, stream>>>(xb, Wqb, bq, Qb,  S, H, H, SCALING);
  gemm_bf16_kernel<0><<<gGrid, gBlk, 0, stream>>>(xb, Wkb, bk, Kbb, S, H, H, 1.0f);
  gemm_bf16_kernel<1><<<gGrid, gBlk, 0, stream>>>(xb, Wvb, bv, Vt,  S, H, H, 1.0f);

  // 3) Causal flash attention
  dim3 aGrid(NHn, S / 64), aBlk(128);
  flash_attn_kernel<<<aGrid, aBlk, 0, stream>>>(Qb, Kbb, Vt, Ab, S, H);

  // 4) Output projection -> fp32 d_out
  gemm_bf16_kernel<2><<<gGrid, gBlk, 0, stream>>>(Ab, Wob, bo, d_out, S, H, H, 1.0f);
}
